// GraphTransformerLayer_48009144434784
// MI455X (gfx1250) — compile-verified, matrix operand order checked by reading, not run on hardware
//
#include <hip/hip_runtime.h>

typedef __attribute__((ext_vector_type(16))) _Float16 v16h;
typedef __attribute__((ext_vector_type(8)))  _Float16 v8h;
typedef __attribute__((ext_vector_type(4)))  _Float16 v4h;
typedef __attribute__((ext_vector_type(8)))  float    v8f;

#define HIDDIM 128
#define NHEAD  8
#define DHEAD  16
#define QKVS   384           // packed Q|K|V row stride

// ---------- order-preserving float<->uint key for atomic max ----------
__device__ __forceinline__ unsigned f2key(float f) {
  unsigned u = __float_as_uint(f);
  return (u & 0x80000000u) ? ~u : (u | 0x80000000u);
}
__device__ __forceinline__ float key2f(unsigned k) {
  return (k & 0x80000000u) ? __uint_as_float(k & 0x7fffffffu) : __uint_as_float(~k);
}

// ---------- weight convert + transpose: W[K][Nc] f32 -> Wt[Nc][K] f16 ----------
__global__ void wt_cvt_kernel(const float* __restrict__ W, _Float16* __restrict__ Wt,
                              int K, int Nc) {
  int t = blockIdx.x * blockDim.x + threadIdx.x;
  if (t >= K * Nc) return;
  int k = t / Nc, c = t - k * Nc;
  Wt[(size_t)c * K + k] = (_Float16)W[t];
}

// ---------- concat three 128-float biases into one 384-float vector ----------
__global__ void concat3_kernel(const float* __restrict__ a, const float* __restrict__ b,
                               const float* __restrict__ c, float* __restrict__ o) {
  int t = threadIdx.x;  // launched with 384 threads, 1 block
  if (t < 128)       o[t] = a[t];
  else if (t < 256)  o[t] = b[t - 128];
  else if (t < 384)  o[t] = c[t - 256];
}

// ---------- elementwise f32 -> f16 ----------
__global__ void f2h_kernel(const float* __restrict__ in, _Float16* __restrict__ out, int n) {
  int t = blockIdx.x * blockDim.x + threadIdx.x;
  if (t < n) out[t] = (_Float16)in[t];
}

// ---------- LayerNorm: one wave (32 lanes) per row of 128 ----------
__global__ void ln_kernel(const float* __restrict__ x, const float* __restrict__ g,
                          const float* __restrict__ b, _Float16* __restrict__ out, int n) {
  int row  = (blockIdx.x * blockDim.x + threadIdx.x) >> 5;
  int lane = threadIdx.x & 31;
  if (row >= n) return;
  const float* xr = x + (size_t)row * HIDDIM;
  float v[4]; float s = 0.f;
#pragma unroll
  for (int i = 0; i < 4; ++i) { v[i] = xr[lane * 4 + i]; s += v[i]; }
#pragma unroll
  for (int m = 16; m >= 1; m >>= 1) s += __shfl_xor(s, m, 32);
  float mu = s * (1.0f / HIDDIM);
  float vs = 0.f;
#pragma unroll
  for (int i = 0; i < 4; ++i) { float d = v[i] - mu; vs += d * d; }
#pragma unroll
  for (int m = 16; m >= 1; m >>= 1) vs += __shfl_xor(vs, m, 32);
  float inv = rsqrtf(vs * (1.0f / HIDDIM) + 1e-5f);
  v4h o;
#pragma unroll
  for (int i = 0; i < 4; ++i) {
    int c = lane * 4 + i;
    o[i] = (_Float16)((v[i] - mu) * inv * g[c] + b[c]);
  }
  *(v4h*)(out + (size_t)row * HIDDIM + lane * 4) = o;
}

// ---------- 16x32 f16 fragment from LDS tile [16 rows][32 k], per ISA 7.12.2 ----------
__device__ __forceinline__ v16h frag_from_lds(const _Float16* base, int lane) {
  int r  = lane & 15;
  int hh = (lane >> 4) << 3;
  const _Float16* p = base + r * 32;
  v8h lo = *(const v8h*)(p + hh);
  v8h hi = *(const v8h*)(p + 16 + hh);
  v16h f;
#pragma unroll
  for (int e = 0; e < 8; ++e) { f[e] = lo[e]; f[e + 8] = hi[e]; }
  return f;
}

// ---------- WMMA GEMM: out[M,Nc] = A[M,K](f16) @ Bt[Nc,K]^T(f16) + bias (+res, relu) ----
// block = 256 threads = 8 waves; block tile = 128 rows x 128 cols; K staged in 32-slices.
// Full blocks: double-buffered LDS fed by global_load_async_to_lds_b128 (ASYNCcnt),
// overlapping the async DMA of slice k+1 with the WMMAs of slice k.
template <bool RELU, bool OUTH>
__global__ __launch_bounds__(256)
void wmma_gemm_kernel(const _Float16* __restrict__ A, const _Float16* __restrict__ Bt,
                      const float* __restrict__ bias, const float* __restrict__ res,
                      float* __restrict__ outF, _Float16* __restrict__ outH,
                      int M, int K, int Nc) {
  __shared__ __align__(16) _Float16 As[2][128 * 32];
  __shared__ __align__(16) _Float16 Bs[2][128 * 32];
  const int m0 = blockIdx.x * 128;
  const int n0 = blockIdx.y * 128;
  const int t = threadIdx.x;
  const int wave = t >> 5, lane = t & 31;
  v8f acc[8] = {};

  const int lr  = t >> 1;          // 0..127: row (A) / col (B) this thread stages
  const int seg = (t & 1) << 4;    // half-offset 0 or 16 within the 32-wide K slice
  // low 32 bits of a generic pointer to __shared__ == LDS byte offset
  const unsigned aLds = (unsigned)(uintptr_t)(&As[0][0] + lr * 32 + seg);
  const unsigned bLds = (unsigned)(uintptr_t)(&Bs[0][0] + lr * 32 + seg);
  const unsigned tileB = 128 * 32 * 2;  // bytes per LDS tile buffer

  auto compute_slice = [&](int buf) {
    v16h af = frag_from_lds(&As[buf][0] + wave * (16 * 32), lane);
#pragma unroll
    for (int ct = 0; ct < 8; ++ct) {
      v16h bf = frag_from_lds(&Bs[buf][0] + ct * (16 * 32), lane);
      acc[ct] = __builtin_amdgcn_wmma_f32_16x16x32_f16(false, af, false, bf,
                                                       (short)0, acc[ct], false, false);
    }
  };

  if (m0 + 128 <= M) {
    // ---- async double-buffered path (full blocks) ----
    auto issue_async = [&](int buf, int kt) {
      const _Float16* ag = A  + (size_t)(m0 + lr) * K + kt + seg;
      const _Float16* bg = Bt + (size_t)(n0 + lr) * K + kt + seg;
      unsigned al = aLds + (unsigned)buf * tileB;
      unsigned bl = bLds + (unsigned)buf * tileB;
      // INST_OFFSET adds to both the LDS and memory address -> two 16B chunks
      asm volatile("global_load_async_to_lds_b128 %0, %1, off"
                   :: "v"(al), "v"(ag) : "memory");
      asm volatile("global_load_async_to_lds_b128 %0, %1, off offset:16"
                   :: "v"(al), "v"(ag) : "memory");
      asm volatile("global_load_async_to_lds_b128 %0, %1, off"
                   :: "v"(bl), "v"(bg) : "memory");
      asm volatile("global_load_async_to_lds_b128 %0, %1, off offset:16"
                   :: "v"(bl), "v"(bg) : "memory");
    };
    issue_async(0, 0);
    int cur = 0;
    for (int kt = 0; kt < K; kt += 32) {
      asm volatile("s_wait_asynccnt 0x0" ::: "memory");  // my tile arrived
      __syncthreads();                                    // everyone's tile arrived / prev reads done
      if (kt + 32 < K) issue_async(cur ^ 1, kt + 32);     // prefetch next slice during compute
      compute_slice(cur);
      cur ^= 1;
    }
  } else {
    // ---- guarded synchronous path (ragged last block only) ----
    for (int kt = 0; kt < K; kt += 32) {
      {
        int gr = m0 + lr;
        v8h a0 = {}, a1 = {};
        if (gr < M) {
          const v8h* src = (const v8h*)(A + (size_t)gr * K + kt + seg);
          a0 = src[0]; a1 = src[1];
        }
        *(v8h*)(&As[0][0] + lr * 32 + seg)     = a0;
        *(v8h*)(&As[0][0] + lr * 32 + seg + 8) = a1;
      }
      {
        const v8h* src = (const v8h*)(Bt + (size_t)(n0 + lr) * K + kt + seg);
        *(v8h*)(&Bs[0][0] + lr * 32 + seg)     = src[0];
        *(v8h*)(&Bs[0][0] + lr * 32 + seg + 8) = src[1];
      }
      __syncthreads();
      compute_slice(0);
      __syncthreads();
    }
  }

  // C layout (ISA): lane 0-15 col=lane rows i+0..7 ; lane 16-31 col=lane-16 rows 8+i
  int r0   = m0 + wave * 16 + ((lane >> 4) << 3);
  int cSub = lane & 15;
#pragma unroll
  for (int ct = 0; ct < 8; ++ct) {
    int col  = n0 + ct * 16 + cSub;
    float bv = bias[col];
#pragma unroll
    for (int i = 0; i < 8; ++i) {
      int row = r0 + i;
      if (row < M) {
        float v = acc[ct][i] + bv;
        if (RELU) v = v > 0.f ? v : 0.f;
        size_t idx = (size_t)row * Nc + col;
        if (res) v += res[idx];
        if (OUTH) outH[idx] = (_Float16)v;
        else      outF[idx] = v;
      }
    }
  }
}

// ---------- edge pass 1: scores[e,h] + global max (QKV packed, stride 384) ----------
__global__ void edge_score_kernel(const int* __restrict__ ei, const int* __restrict__ ea,
                                  const float* __restrict__ QKV,
                                  const float* __restrict__ ebias,
                                  float* __restrict__ scores, unsigned* __restrict__ gmax,
                                  int E) {
  int t = blockIdx.x * blockDim.x + threadIdx.x;
  float s = -3.4e38f;
  if (t < E * NHEAD) {
    int e = t >> 3, h = t & 7;
    int src = ei[e], dst = ei[E + e];
    const float* q = QKV + (size_t)dst * QKVS + h * DHEAD;          // Q block
    const float* k = QKV + (size_t)src * QKVS + HIDDIM + h * DHEAD; // K block
    float d = 0.f;
#pragma unroll
    for (int i = 0; i < DHEAD; ++i) d += q[i] * k[i];
    int a = ea[e]; a = a < 0 ? 0 : (a > 12 ? 12 : a);
    s = d * 0.25f + ebias[a * NHEAD + h];   // 1/sqrt(16) = 0.25
    scores[t] = s;
  }
#pragma unroll
  for (int m = 16; m >= 1; m >>= 1) { float o = __shfl_xor(s, m, 32); s = s > o ? s : o; }
  if ((threadIdx.x & 31) == 0) atomicMax(gmax, f2key(s));
}

// ---------- edge pass 2: exp + segment sum over dst ----------
__global__ void edge_exp_kernel(const int* __restrict__ ei, float* __restrict__ scores,
                                float* __restrict__ asum, const unsigned* __restrict__ gmax,
                                int E) {
  int t = blockIdx.x * blockDim.x + threadIdx.x;
  if (t >= E * NHEAD) return;
  int e = t >> 3, h = t & 7;
  int dst = ei[E + e];
  float es = expf(scores[t] - key2f(*gmax));
  scores[t] = es;
  atomicAdd(&asum[dst * NHEAD + h], es);
}

// ---------- edge pass 3: weighted message scatter (V at offset 256 in QKV row) ------
__global__ void edge_msg_kernel(const int* __restrict__ ei, const float* __restrict__ scores,
                                const float* __restrict__ asum, const float* __restrict__ QKV,
                                float* __restrict__ attnout, int E) {
  long long t = (long long)blockIdx.x * blockDim.x + threadIdx.x;
  if (t >= (long long)E * HIDDIM) return;
  int e = (int)(t >> 7);
  int c = (int)(t & 127);
  int h = c >> 4;
  int src = ei[e], dst = ei[E + e];
  float w = scores[e * NHEAD + h] / (asum[dst * NHEAD + h] + 1e-8f);
  float vv = QKV[(size_t)src * QKVS + 2 * HIDDIM + c];
  atomicAdd(&attnout[(size_t)dst * HIDDIM + c], vv * w);
}

extern "C" void kernel_launch(void* const* d_in, const int* in_sizes, int n_in,
                              void* d_out, int out_size, void* d_ws, size_t ws_size,
                              hipStream_t stream) {
  const float* x     = (const float*)d_in[0];
  const int*   ei    = (const int*)d_in[1];
  const int*   ea    = (const int*)d_in[2];
  const float* Wq    = (const float*)d_in[3];
  const float* bq    = (const float*)d_in[4];
  const float* Wk    = (const float*)d_in[5];
  const float* bk    = (const float*)d_in[6];
  const float* Wv    = (const float*)d_in[7];
  const float* bv    = (const float*)d_in[8];
  const float* Wo    = (const float*)d_in[9];
  const float* bo    = (const float*)d_in[10];
  const float* ebias = (const float*)d_in[11];
  const float* W1    = (const float*)d_in[12];
  const float* b1    = (const float*)d_in[13];
  const float* W2    = (const float*)d_in[14];
  const float* b2    = (const float*)d_in[15];
  const float* g1    = (const float*)d_in[16];
  const float* be1   = (const float*)d_in[17];
  const float* g2    = (const float*)d_in[18];
  const float* be2   = (const float*)d_in[19];
  float* out = (float*)d_out;

  const int N = in_sizes[0] / HIDDIM;
  const int E = in_sizes[2];

  // ---- workspace layout (256B aligned slots, lifetime-based reuse) ----
  char* ws = (char*)d_ws;
  size_t off = 0;
  auto take = [&](size_t bytes) -> void* {
    void* p = ws + off;
    off = (off + bytes + 255) & ~(size_t)255;
    return p;
  };
  _Float16* WqkvT = (_Float16*)take((size_t)QKVS * 128 * 2);  // [384][128] transposed
  _Float16* WoT   = (_Float16*)take((size_t)128 * 128 * 2);
  _Float16* W1T   = (_Float16*)take((size_t)512 * 128 * 2);
  _Float16* W2T   = (_Float16*)take((size_t)128 * 512 * 2);
  float*    bqkv  = (float*)take((size_t)QKVS * 4);
  unsigned* gmax  = (unsigned*)take(256);
  float*    asum  = (float*)take((size_t)N * NHEAD * 4);
  _Float16* h1h   = (_Float16*)take((size_t)N * HIDDIM * 2);
  float*    QKV   = (float*)take((size_t)N * QKVS * 4);
  float*    scores  = (float*)take((size_t)E * NHEAD * 4);
  float*    attnout = (float*)take((size_t)N * HIDDIM * 4);
  // lifetime-safe aliases:
  _Float16* aouth = h1h;                                   // h1 dead after QKV gemm
  float*    x2    = QKV;                                   // QKV dead after pass3
  _Float16* h2h   = (_Float16*)(QKV + (size_t)N * HIDDIM); // disjoint from x2 bytes
  _Float16* ffn1h = (_Float16*)scores;                     // E*8*4 >= N*512*2

  // 1) convert + transpose weights to f16 [Nc][K]; pack Q|K|V
  wt_cvt_kernel<<<(128 * 128 + 255) / 256, 256, 0, stream>>>(Wq, WqkvT, 128, 128);
  wt_cvt_kernel<<<(128 * 128 + 255) / 256, 256, 0, stream>>>(Wk, WqkvT + 128 * 128, 128, 128);
  wt_cvt_kernel<<<(128 * 128 + 255) / 256, 256, 0, stream>>>(Wv, WqkvT + 2 * 128 * 128, 128, 128);
  wt_cvt_kernel<<<(128 * 128 + 255) / 256, 256, 0, stream>>>(Wo, WoT, 128, 128);
  wt_cvt_kernel<<<(128 * 512 + 255) / 256, 256, 0, stream>>>(W1, W1T, 128, 512);
  wt_cvt_kernel<<<(512 * 128 + 255) / 256, 256, 0, stream>>>(W2, W2T, 512, 128);
  concat3_kernel<<<1, QKVS, 0, stream>>>(bq, bk, bv, bqkv);

  // 2) LN1 -> f16
  ln_kernel<<<(N * 32 + 255) / 256, 256, 0, stream>>>(x, g1, be1, h1h, N);

  // 3) fused Q|K|V projection (WMMA, Nc=384)
  dim3 gQKV((N + 127) / 128, 3);
  wmma_gemm_kernel<false, false><<<gQKV, 256, 0, stream>>>(h1h, WqkvT, bqkv, nullptr,
                                                           QKV, nullptr, N, 128, QKVS);

  // 4) attention edge passes
  hipMemsetAsync(gmax, 0, 4, stream);
  hipMemsetAsync(asum, 0, (size_t)N * NHEAD * 4, stream);
  hipMemsetAsync(attnout, 0, (size_t)N * HIDDIM * 4, stream);
  edge_score_kernel<<<(E * NHEAD + 255) / 256, 256, 0, stream>>>(ei, ea, QKV, ebias, scores, gmax, E);
  edge_exp_kernel<<<(E * NHEAD + 255) / 256, 256, 0, stream>>>(ei, scores, asum, gmax, E);
  long long tot3 = (long long)E * HIDDIM;
  edge_msg_kernel<<<(unsigned)((tot3 + 255) / 256), 256, 0, stream>>>(ei, scores, asum, QKV, attnout, E);

  // 5) output projection + residual -> x2 (reuses QKV storage, dead after pass3)
  f2h_kernel<<<(N * HIDDIM + 255) / 256, 256, 0, stream>>>(attnout, aouth, N * HIDDIM);
  dim3 g128((N + 127) / 128, 1);
  wmma_gemm_kernel<false, false><<<g128, 256, 0, stream>>>(aouth, WoT, bo, x, x2, nullptr, N, 128, 128);

  // 6) LN2 + FFN
  ln_kernel<<<(N * 32 + 255) / 256, 256, 0, stream>>>(x2, g2, be2, h2h, N);
  dim3 gF1((N + 127) / 128, 4);
  wmma_gemm_kernel<true, true><<<gF1, 256, 0, stream>>>(h2h, W1T, b1, nullptr, nullptr, ffn1h, N, 128, 512);
  wmma_gemm_kernel<false, false><<<g128, 256, 0, stream>>>(ffn1h, W2T, b2, x2, out, nullptr, N, 512, 128);
}
